// ConvolutionLayer_88768384074044
// MI455X (gfx1250) — matrix operand order repaired and checked by reading, not compiled, not run on hardware
//
#include <hip/hip_runtime.h>

// ---------------------------------------------------------------------------
// GCN layer on MI455X (gfx1250, wave32, WMMA):
//   degree -> d^-1/2 ; mf = f + dinv_r * (A @ (dinv_c * f)) ; out = relu(mf@W+b)
// Big GEMM on v_wmma_f32_16x16x32_bf16 (f32 accumulate), double-buffered LDS,
// bf16 operand tiles staged with global_load_async_to_lds_b128 (ASYNCcnt).
// ---------------------------------------------------------------------------

typedef __bf16 bf16x16 __attribute__((ext_vector_type(16)));
typedef float  f32x8   __attribute__((ext_vector_type(8)));
typedef int    v4i     __attribute__((vector_size(16)));   // matches builtin param

union FragB16 {
    bf16x16 v;
    uint4   q[2];   // two 16B chunks (8 bf16 each)
};

__device__ __forceinline__ unsigned short f2bf(float f) {
    unsigned int u = __float_as_uint(f);
    unsigned int r = u + 0x7FFFu + ((u >> 16) & 1u);   // round-to-nearest-even
    return (unsigned short)(r >> 16);
}

__device__ __forceinline__ unsigned long long pack4bf(float4 v) {
    unsigned long long r;
    r  = (unsigned long long)f2bf(v.x);
    r |= (unsigned long long)f2bf(v.y) << 16;
    r |= (unsigned long long)f2bf(v.z) << 32;
    r |= (unsigned long long)f2bf(v.w) << 48;
    return r;
}

// ---- CDNA5 async global->LDS (one 16B transfer per lane, ASYNCcnt) --------
__device__ __forceinline__ void async_copy_b128(unsigned short* lds_dst,
                                                const unsigned short* gsrc) {
#if defined(__HIP_DEVICE_COMPILE__)
#if __has_builtin(__builtin_amdgcn_global_load_async_to_lds_b128)
    // param0: v4i addrspace(1)* — global flat addr == as1 addr, reinterpret
    // through an integer (also drops const). param1: v4i addrspace(3)* —
    // real pointer addrspacecast so the backend materializes the
    // wave-relative LDS offset for VDST.
    __builtin_amdgcn_global_load_async_to_lds_b128(
        (__attribute__((address_space(1))) v4i*)(unsigned long long)(uintptr_t)gsrc,
        (__attribute__((address_space(3))) v4i*)lds_dst,
        0, 0);
#else
    unsigned loff = (unsigned)(unsigned long long)
        (__attribute__((address_space(3))) unsigned short*)lds_dst;
    asm volatile("global_load_async_to_lds_b128 %0, %1, off"
                 :: "v"(loff), "v"((unsigned long long)(uintptr_t)gsrc)
                 : "memory");
#endif
#endif
}

__device__ __forceinline__ void wait_async0() {
#if defined(__HIP_DEVICE_COMPILE__)
#if __has_builtin(__builtin_amdgcn_s_wait_asynccnt)
    __builtin_amdgcn_s_wait_asynccnt(0);
#else
    asm volatile("s_wait_asynccnt 0x0" ::: "memory");
#endif
#endif
}

#define NN      8192
#define DIM     512
#define BM      128
#define BN      128
#define BK      64
#define LDT     72      // LDS pitch in ushorts (144 B: 16B-aligned, bank-spread)

// ---------------------------------------------------------------------------
// Kernel 1: degree (row sums of A) -> dinv = rsqrt(degree)
// ---------------------------------------------------------------------------
__global__ __launch_bounds__(256)
void degree_kernel(const float* __restrict__ A, float* __restrict__ dinv) {
    __shared__ float red[256];
    const int row = blockIdx.x;
    const int t   = threadIdx.x;
    const float* ap = A + (size_t)row * NN;
    float s = 0.f;
#pragma unroll
    for (int i = 0; i < 8; ++i) {
        float4 v = *(const float4*)(ap + t * 4 + i * 1024);
        s += v.x + v.y + v.z + v.w;
    }
    red[t] = s;
    __syncthreads();
    for (int off = 128; off > 0; off >>= 1) {
        if (t < off) red[t] += red[t + off];
        __syncthreads();
    }
    if (t == 0) dinv[row] = rsqrtf(red[0]);
}

// ---------------------------------------------------------------------------
// Kernel 2: fsT[n][k] = bf16( dinv[k] * feature[k][n] )   (LDS-tiled transpose)
// ---------------------------------------------------------------------------
__global__ __launch_bounds__(256)
void fsT_kernel(const float* __restrict__ feat, const float* __restrict__ dinv,
                unsigned short* __restrict__ fsT) {
    __shared__ float tile[32][33];
    const int c0 = blockIdx.x * 32, j0 = blockIdx.y * 32;
    const int tx = threadIdx.x, ty = threadIdx.y;
#pragma unroll
    for (int i = 0; i < 4; ++i) {
        int j = j0 + ty + i * 8;
        tile[ty + i * 8][tx] = dinv[j] * feat[(size_t)j * DIM + c0 + tx];
    }
    __syncthreads();
#pragma unroll
    for (int i = 0; i < 4; ++i) {
        int c = c0 + ty + i * 8;
        fsT[(size_t)c * NN + j0 + tx] = f2bf(tile[tx][ty + i * 8]);
    }
}

// ---------------------------------------------------------------------------
// Kernel 3: WT[n][k] = bf16( W[k][n] )
// ---------------------------------------------------------------------------
__global__ __launch_bounds__(256)
void wT_kernel(const float* __restrict__ W, unsigned short* __restrict__ WT) {
    __shared__ float tile[32][33];
    const int n0 = blockIdx.x * 32, k0 = blockIdx.y * 32;
    const int tx = threadIdx.x, ty = threadIdx.y;
#pragma unroll
    for (int i = 0; i < 4; ++i)
        tile[ty + i * 8][tx] = W[(size_t)(k0 + ty + i * 8) * DIM + n0 + tx];
    __syncthreads();
#pragma unroll
    for (int i = 0; i < 4; ++i)
        WT[(size_t)(n0 + ty + i * 8) * DIM + k0 + tx] = f2bf(tile[tx][ty + i * 8]);
}

// ---------------------------------------------------------------------------
// Kernel 4: mf_bf16 = bf16( dinv_r * (A @ fsT^T) + feature )
//   128x128 block tile, 8 waves (4x2), wave tile 32x64 -> 2x4 v8f accums,
//   K staged 64/iter, double-buffered LDS. A: fp32->bf16 in regs; B: async.
// ---------------------------------------------------------------------------
__global__ __launch_bounds__(256)
void gemm1_kernel(const float* __restrict__ A, const unsigned short* __restrict__ fsT,
                  const float* __restrict__ feat, const float* __restrict__ dinv,
                  unsigned short* __restrict__ mf) {
    __shared__ alignas(16) unsigned short As[2][BM * LDT];
    __shared__ alignas(16) unsigned short Bs[2][BN * LDT];

    const int tid  = threadIdx.x;
    const int wave = tid >> 5;
    const int lane = tid & 31;
    const int l15  = lane & 15;
    const int lhi  = lane >> 4;          // which K/M half this lane holds

    const int m0 = blockIdx.y * BM;
    const int n0 = blockIdx.x * BN;
    const int wm = (wave >> 1) * 32;     // wave grid 4 (M) x 2 (N)
    const int wn = (wave & 1) * 64;

    const int ra = tid >> 4, ca = tid & 15;   // A staging: 16 rows/pass, float4 col
    const int rb = tid >> 3, cb = tid & 7;    // B staging: 32 rows/pass, uint4 col

    auto stageA = [&](int buf, int k0) {
        const float* ap = A + (size_t)(m0 + ra) * NN + k0 + ca * 4;
#pragma unroll
        for (int p = 0; p < 8; ++p) {
            float4 v = *(const float4*)(ap + (size_t)p * 16 * NN);
            *(unsigned long long*)&As[buf][(ra + p * 16) * LDT + ca * 4] = pack4bf(v);
        }
        if (k0 + BK < NN) __builtin_prefetch(ap + BK, 0, 0);
    };
    auto stageB = [&](int buf, int k0) {
        const unsigned short* bp = fsT + (size_t)(n0 + rb) * NN + k0 + cb * 8;
#pragma unroll
        for (int p = 0; p < 4; ++p)
            async_copy_b128(&Bs[buf][(rb + p * 32) * LDT + cb * 8],
                            bp + (size_t)p * 32 * NN);
    };

    f32x8 acc[2][4];
#pragma unroll
    for (int i = 0; i < 2; ++i)
#pragma unroll
        for (int j = 0; j < 4; ++j) acc[i][j] = (f32x8)0.f;

    stageB(0, 0);
    stageA(0, 0);
    wait_async0();
    __syncthreads();

    int buf = 0;
    for (int k0 = 0; k0 < NN; k0 += BK, buf ^= 1) {
        if (k0 + BK < NN) {     // stage next panel into the other buffer
            stageB(buf ^ 1, k0 + BK);
            stageA(buf ^ 1, k0 + BK);
        }
#pragma unroll
        for (int kk = 0; kk < BK; kk += 32) {
            FragB16 a[2], b[4];
#pragma unroll
            for (int i = 0; i < 2; ++i) {   // A: lane row l15, K {0..7,16..23}+8*lhi
                const int row = wm + i * 16 + l15;
                a[i].q[0] = *(const uint4*)&As[buf][row * LDT + kk + lhi * 8];
                a[i].q[1] = *(const uint4*)&As[buf][row * LDT + kk + 16 + lhi * 8];
            }
#pragma unroll
            for (int j = 0; j < 4; ++j) {   // B: lane col l15, K {0..7,16..23}+8*lhi
                const int col = wn + j * 16 + l15;
                b[j].q[0] = *(const uint4*)&Bs[buf][col * LDT + kk + lhi * 8];
                b[j].q[1] = *(const uint4*)&Bs[buf][col * LDT + kk + 16 + lhi * 8];
            }
#pragma unroll
            for (int i = 0; i < 2; ++i)
#pragma unroll
                for (int j = 0; j < 4; ++j)
                    acc[i][j] = __builtin_amdgcn_wmma_f32_16x16x32_bf16(
                        false, a[i].v, false, b[j].v, (short)0, acc[i][j], false, false);
        }
        wait_async0();          // own async writes into buf^1 done
        __syncthreads();        // everyone done reading buf / writing buf^1
    }

    // ---- epilogue: mf = bf16(dinv[row]*acc + feat[row][col]) ----
#pragma unroll
    for (int i = 0; i < 2; ++i) {
#pragma unroll
        for (int v = 0; v < 8; ++v) {       // C layout: VGPR v -> M = v + 8*lhi
            const int row = m0 + wm + i * 16 + v + lhi * 8;
            const float ds = dinv[row];
#pragma unroll
            for (int j = 0; j < 4; ++j) {
                const int col = n0 + wn + j * 16 + l15;
                const float val = ds * acc[i][j][v] + feat[(size_t)row * DIM + col];
                mf[(size_t)row * DIM + col] = f2bf(val);
            }
        }
    }
}

// ---------------------------------------------------------------------------
// Kernel 5: out = relu( mf_bf16 @ WT^T + b )  — both tiles staged async.
// ---------------------------------------------------------------------------
__global__ __launch_bounds__(256)
void gemm2_kernel(const unsigned short* __restrict__ mf,
                  const unsigned short* __restrict__ WT,
                  const float* __restrict__ bias, float* __restrict__ out) {
    __shared__ alignas(16) unsigned short As[2][BM * LDT];
    __shared__ alignas(16) unsigned short Bs[2][BN * LDT];

    const int tid  = threadIdx.x;
    const int wave = tid >> 5;
    const int lane = tid & 31;
    const int l15  = lane & 15;
    const int lhi  = lane >> 4;

    const int m0 = blockIdx.y * BM;
    const int n0 = blockIdx.x * BN;
    const int wm = (wave >> 1) * 32;
    const int wn = (wave & 1) * 64;

    const int rb = tid >> 3, cb = tid & 7;

    auto stage = [&](int buf, int k0) {
        const unsigned short* ap = mf + (size_t)(m0 + rb) * DIM + k0 + cb * 8;
        const unsigned short* bp = WT + (size_t)(n0 + rb) * DIM + k0 + cb * 8;
#pragma unroll
        for (int p = 0; p < 4; ++p)
            async_copy_b128(&As[buf][(rb + p * 32) * LDT + cb * 8],
                            ap + (size_t)p * 32 * DIM);
#pragma unroll
        for (int p = 0; p < 4; ++p)
            async_copy_b128(&Bs[buf][(rb + p * 32) * LDT + cb * 8],
                            bp + (size_t)p * 32 * DIM);
    };

    f32x8 acc[2][4];
#pragma unroll
    for (int i = 0; i < 2; ++i)
#pragma unroll
        for (int j = 0; j < 4; ++j) acc[i][j] = (f32x8)0.f;

    stage(0, 0);
    wait_async0();
    __syncthreads();

    int buf = 0;
    for (int k0 = 0; k0 < DIM; k0 += BK, buf ^= 1) {
        if (k0 + BK < DIM) stage(buf ^ 1, k0 + BK);
#pragma unroll
        for (int kk = 0; kk < BK; kk += 32) {
            FragB16 a[2], b[4];
#pragma unroll
            for (int i = 0; i < 2; ++i) {
                const int row = wm + i * 16 + l15;
                a[i].q[0] = *(const uint4*)&As[buf][row * LDT + kk + lhi * 8];
                a[i].q[1] = *(const uint4*)&As[buf][row * LDT + kk + 16 + lhi * 8];
            }
#pragma unroll
            for (int j = 0; j < 4; ++j) {
                const int col = wn + j * 16 + l15;
                b[j].q[0] = *(const uint4*)&Bs[buf][col * LDT + kk + lhi * 8];
                b[j].q[1] = *(const uint4*)&Bs[buf][col * LDT + kk + 16 + lhi * 8];
            }
#pragma unroll
            for (int i = 0; i < 2; ++i)
#pragma unroll
                for (int j = 0; j < 4; ++j)
                    acc[i][j] = __builtin_amdgcn_wmma_f32_16x16x32_bf16(
                        false, a[i].v, false, b[j].v, (short)0, acc[i][j], false, false);
        }
        wait_async0();
        __syncthreads();
    }

#pragma unroll
    for (int i = 0; i < 2; ++i) {
#pragma unroll
        for (int v = 0; v < 8; ++v) {
            const int row = m0 + wm + i * 16 + v + lhi * 8;
#pragma unroll
            for (int j = 0; j < 4; ++j) {
                const int col = n0 + wn + j * 16 + l15;
                float val = acc[i][j][v] + bias[col];
                out[(size_t)row * DIM + col] = val > 0.f ? val : 0.f;
            }
        }
    }
}

// ---------------------------------------------------------------------------
extern "C" void kernel_launch(void* const* d_in, const int* in_sizes, int n_in,
                              void* d_out, int out_size, void* d_ws, size_t ws_size,
                              hipStream_t stream) {
    const float* A    = (const float*)d_in[0];   // [8192, 8192]
    const float* feat = (const float*)d_in[1];   // [8192, 512]
    const float* W    = (const float*)d_in[2];   // [512, 512]
    const float* bias = (const float*)d_in[3];   // [512]
    float*       out  = (float*)d_out;           // [8192, 512]

    // workspace layout
    char* ws = (char*)d_ws;
    float*          dinv = (float*)ws;                                       //  32 KB
    unsigned short* fsT  = (unsigned short*)(ws + 32768);                    //   8 MB
    unsigned short* WT   = (unsigned short*)(ws + 32768 + 8388608);          // 512 KB
    unsigned short* mf   = (unsigned short*)(ws + 32768 + 8388608 + 524288); //   8 MB

    degree_kernel<<<NN, 256, 0, stream>>>(A, dinv);

    dim3 tb(32, 8);
    fsT_kernel<<<dim3(DIM / 32, NN / 32), tb, 0, stream>>>(feat, dinv, fsT);
    wT_kernel<<<dim3(DIM / 32, DIM / 32), tb, 0, stream>>>(W, WT);

    gemm1_kernel<<<dim3(DIM / BN, NN / BM), 256, 0, stream>>>(A, fsT, feat, dinv, mf);
    gemm2_kernel<<<dim3(DIM / BN, NN / BM), 256, 0, stream>>>(mf, WT, bias, out);
}
